// TransformerBlock_2697239461940
// MI455X (gfx1250) — matrix operand order repaired and checked by reading, not compiled, not run on hardware
//
#include <hip/hip_runtime.h>
#include <hip/hip_bf16.h>
#include <math.h>

typedef __attribute__((ext_vector_type(16))) _Float16 v16h;
typedef __attribute__((ext_vector_type(8)))  float    v8f;

#define DM   1024      // d_model
#define NH   16        // heads
#define DK   64        // head dim
#define DFF  4096      // ffn dim
#define SQ   2048      // seq len
#define BB   2         // batch
#define TT   (BB*SQ)   // tokens = 4096

// ---------------------------------------------------------------------------
// CDNA5 async copy: global -> LDS, 16B per lane, tracked by ASYNCcnt.
// ---------------------------------------------------------------------------
static __device__ inline void async_fill_b128(unsigned lds_off, const void* gaddr) {
    asm volatile("global_load_async_to_lds_b128 %0, %1, off"
                 :: "v"(lds_off), "v"(gaddr) : "memory");
}
static __device__ inline void wait_async_le1() {
    asm volatile("s_wait_asynccnt 0x1" ::: "memory");
}
static __device__ inline void wait_async_0() {
    asm volatile("s_wait_asynccnt 0x0" ::: "memory");
}

// ---------------------------------------------------------------------------
// WMMA fragment loaders (layouts per CDNA5 ISA §7.12.2, wave32)
// ---------------------------------------------------------------------------

// A: 16x32 f16, row-major source with leading dim lda (in elements).
// lane 0-15: M=lane, K-halves {base..base+7, base+16..base+23}, base=0
// lane 16-31: same rows, base=8.
static __device__ inline v16h load_a16(const _Float16* A, int lda, int lane) {
    const int m    = lane & 15;
    const int base = (lane >> 4) << 3;
    const _Float16* r = A + (size_t)m * lda + base;
    v16h a;
    ((float4*)&a)[0] = *((const float4*)r);          // K = base .. base+7
    ((float4*)&a)[1] = *((const float4*)(r + 16));   // K = base+16 .. base+23
    return a;
}

// Same A fragment but from an f32 source (attention probabilities), convert.
static __device__ inline v16h load_a32(const float* A, int lda, int lane) {
    const int m    = lane & 15;
    const int base = (lane >> 4) << 3;
    const float* r = A + (size_t)m * lda + base;
    v16h a;
#pragma unroll
    for (int j = 0; j < 8; ++j) {
        a[j]     = (_Float16)r[j];
        a[8 + j] = (_Float16)r[16 + j];
    }
    return a;
}

// B: 32x16 f16 fragment from the LDS-staged 32x64 tile (row stride 64 halves).
// lane: N = lane&15; elements K = {base..base+7, base+16..base+23}, base = 8*(lane>>4).
static __device__ inline v16h load_b_lds(const _Float16* Bp, int lane) {
    const int n    = lane & 15;
    const int base = (lane >> 4) << 3;
    v16h b;
#pragma unroll
    for (int j = 0; j < 8; ++j) {
        b[j]     = Bp[(size_t)(base + j) * 64 + n];
        b[8 + j] = Bp[(size_t)(base + 16 + j) * 64 + n];
    }
    return b;
}

// ---------------------------------------------------------------------------
// Block-level GEMM inner loop. Block = 256 threads (8 waves),
// tile 128(M) x 64(N), K-step 32.
// B tile (32x64 f16 = 4KB) is staged into LDS with async copies,
// double-buffered: fill of tile i+1 overlaps WMMAs on tile i.
// Each 256-thread block issues 16B/lane => one async op per wave per fill.
// Ap points at (row m0, k=0); Bp points at (k=0, col n0); sB = 2x4KB LDS.
// ---------------------------------------------------------------------------
template <bool A32>
static __device__ inline void block_gemm(const void* Ap, int lda,
                                         const _Float16* Bp, int ldb,
                                         int K, _Float16* sB, v8f acc[4]) {
    const int tid  = threadIdx.x;
    const int lane = tid & 31;
    const int row  = tid >> 3;        // K row within tile: 0..31
    const int col8 = (tid & 7) * 8;   // 16B chunk start within row (halves)

    // stage buffer 0 (k0 = 0)
    async_fill_b128((unsigned)(uintptr_t)(sB + row * 64 + col8),
                    Bp + (size_t)row * ldb + col8);

    const int nIter = K >> 5;
    for (int i = 0; i < nIter; ++i) {
        if (i + 1 < nIter) {
            // prefetch next tile into the other buffer, then wait for current
            async_fill_b128((unsigned)(uintptr_t)(sB + ((i + 1) & 1) * 2048 + row * 64 + col8),
                            Bp + (size_t)((i + 1) * 32 + row) * ldb + col8);
            wait_async_le1();
        } else {
            wait_async_0();
        }
        __syncthreads();   // all waves' async fills of this buffer visible

        const _Float16* bb = sB + (i & 1) * 2048;
        v16h a;
        if (A32) a = load_a32((const float*)Ap + i * 32, lda, lane);
        else     a = load_a16((const _Float16*)Ap + i * 32, lda, lane);
#pragma unroll
        for (int t = 0; t < 4; ++t) {
            v16h b = load_b_lds(bb + 16 * t, lane);
            acc[t] = __builtin_amdgcn_wmma_f32_16x16x32_f16(
                false, a, false, b, (short)0, acc[t], false, false);
        }
        __syncthreads();   // done reading; next iter may overwrite this buffer
    }
}

// ---------------------------------------------------------------------------
// Generic projection GEMM: C[M,N] = A[M,K] * B[K,N] (+bias)(+gelu)(+residual)
// SMODE: 0 = f16 store, 1 = f32 store, 2 = f16 scatter to kT[b,h,dk,s]
// ---------------------------------------------------------------------------
template <int SMODE, bool GELU, bool RESID>
__global__ void gemm_kernel(const _Float16* __restrict__ A, int lda,
                            const _Float16* __restrict__ B, int ldb,
                            void* __restrict__ C, int ldc,
                            const float* __restrict__ bias,
                            const float* __restrict__ resid,
                            int K) {
    __shared__ _Float16 sB[2 * 32 * 64];
    const int lane = threadIdx.x & 31;
    const int w    = threadIdx.x >> 5;
    const int m0   = blockIdx.y * 128 + w * 16;
    const int n0   = blockIdx.x * 64;

    v8f acc[4] = {};
    block_gemm<false>(A + (size_t)m0 * lda, lda, B + n0, ldb, K, sB, acc);

    const int ln = lane & 15;
    const int mo = (lane >> 4) << 3;
#pragma unroll
    for (int t = 0; t < 4; ++t) {
#pragma unroll
        for (int r = 0; r < 8; ++r) {
            const int m = m0 + mo + r;
            const int n = n0 + t * 16 + ln;
            float v = acc[t][r];
            if (bias) v += bias[n];
            if (GELU) v = 0.5f * v * (1.0f + erff(v * 0.70710678118654752f));
            if (RESID) v += resid[(size_t)m * ldc + n];
            if (SMODE == 0) {
                ((_Float16*)C)[(size_t)m * ldc + n] = (_Float16)v;
            } else if (SMODE == 1) {
                ((float*)C)[(size_t)m * ldc + n] = v;
            } else {
                const int b = m / SQ, s = m % SQ, h = n / DK, dk = n % DK;
                ((_Float16*)C)[(((size_t)b * NH + h) * DK + dk) * SQ + s] = (_Float16)v;
            }
        }
    }
}

// ---------------------------------------------------------------------------
// Batched attention scores: per (b,h): scores = (q @ kT) * scale -> f32
// q16: [T, DM] token-major f16; kT: [B,H,DK,SQ] f16; attn: [B,H,SQ,SQ] f32
// ---------------------------------------------------------------------------
__global__ void scores_kernel(const _Float16* __restrict__ q16,
                              const _Float16* __restrict__ kT,
                              float* __restrict__ attn, float scale) {
    __shared__ _Float16 sB[2 * 32 * 64];
    const int lane = threadIdx.x & 31;
    const int w    = threadIdx.x >> 5;
    const int bh   = blockIdx.z;
    const int b    = bh >> 4;
    const int h    = bh & 15;
    const int m0   = blockIdx.y * 128 + w * 16;
    const int n0   = blockIdx.x * 64;

    const _Float16* A  = q16 + ((size_t)b * SQ + m0) * DM + h * DK;
    const _Float16* Bm = kT + (size_t)bh * DK * SQ + n0;

    v8f acc[4] = {};
    block_gemm<false>(A, DM, Bm, SQ, DK, sB, acc);

    float* Cb = attn + (size_t)bh * SQ * SQ;
    const int ln = lane & 15;
    const int mo = (lane >> 4) << 3;
#pragma unroll
    for (int t = 0; t < 4; ++t)
#pragma unroll
        for (int r = 0; r < 8; ++r)
            Cb[(size_t)(m0 + mo + r) * SQ + (n0 + t * 16 + ln)] = acc[t][r] * scale;
}

// ---------------------------------------------------------------------------
// Batched context: per (b,h): ctx = attn(f32->f16) @ v -> f16 token-major
// ---------------------------------------------------------------------------
__global__ void ctx_kernel(const float* __restrict__ attn,
                           const _Float16* __restrict__ vv,
                           _Float16* __restrict__ ctx) {
    __shared__ _Float16 sB[2 * 32 * 64];
    const int lane = threadIdx.x & 31;
    const int w    = threadIdx.x >> 5;
    const int bh   = blockIdx.z;
    const int b    = bh >> 4;
    const int h    = bh & 15;
    const int m0   = blockIdx.y * 128 + w * 16;
    const int n0   = blockIdx.x * 64;   // gridDim.x == 1, N = DK

    const float* A = attn + (size_t)bh * SQ * SQ + (size_t)m0 * SQ;
    const _Float16* Bm = vv + (size_t)b * SQ * DM + h * DK + n0;

    v8f acc[4] = {};
    block_gemm<true>(A, SQ, Bm, DM, SQ, sB, acc);

    const int ln = lane & 15;
    const int mo = (lane >> 4) << 3;
#pragma unroll
    for (int t = 0; t < 4; ++t)
#pragma unroll
        for (int r = 0; r < 8; ++r) {
            const int m = m0 + mo + r;
            const int n = n0 + t * 16 + ln;
            ctx[((size_t)b * SQ + m) * DM + h * DK + n] = (_Float16)acc[t][r];
        }
}

// ---------------------------------------------------------------------------
// LayerNorm (reference: /(std + eps), biased std) -> f16 output
// one block per token, 256 threads, D = 1024 (4 elems/thread)
// ---------------------------------------------------------------------------
__global__ void layernorm_kernel(const float* __restrict__ x,
                                 const float* __restrict__ gamma,
                                 const float* __restrict__ beta,
                                 _Float16* __restrict__ out) {
    __shared__ float red[256];
    const int tid = threadIdx.x;
    const float* xr = x + (size_t)blockIdx.x * DM;

    float v[4], s = 0.0f;
#pragma unroll
    for (int i = 0; i < 4; ++i) { v[i] = xr[tid + 256 * i]; s += v[i]; }
    red[tid] = s; __syncthreads();
    for (int o = 128; o > 0; o >>= 1) {
        if (tid < o) red[tid] += red[tid + o];
        __syncthreads();
    }
    const float mean = red[0] * (1.0f / DM);
    __syncthreads();

    float s2 = 0.0f;
#pragma unroll
    for (int i = 0; i < 4; ++i) { const float d = v[i] - mean; s2 += d * d; }
    red[tid] = s2; __syncthreads();
    for (int o = 128; o > 0; o >>= 1) {
        if (tid < o) red[tid] += red[tid + o];
        __syncthreads();
    }
    const float stdv = sqrtf(red[0] * (1.0f / DM));
    const float inv  = 1.0f / (stdv + 1e-6f);

    _Float16* orow = out + (size_t)blockIdx.x * DM;
#pragma unroll
    for (int i = 0; i < 4; ++i) {
        const int c = tid + 256 * i;
        orow[c] = (_Float16)(gamma[c] * (v[i] - mean) * inv + beta[c]);
    }
}

// ---------------------------------------------------------------------------
// Row softmax in place over SQ=2048 f32 elements; one block per row.
// ---------------------------------------------------------------------------
__global__ void softmax_kernel(float* __restrict__ attn) {
    __shared__ float red[256];
    const int tid = threadIdx.x;
    float* row = attn + (size_t)blockIdx.x * SQ;

    float v[8], mx = -3.0e38f;
#pragma unroll
    for (int i = 0; i < 8; ++i) { v[i] = row[tid + 256 * i]; mx = fmaxf(mx, v[i]); }
    red[tid] = mx; __syncthreads();
    for (int o = 128; o > 0; o >>= 1) {
        if (tid < o) red[tid] = fmaxf(red[tid], red[tid + o]);
        __syncthreads();
    }
    mx = red[0]; __syncthreads();

    float s = 0.0f;
#pragma unroll
    for (int i = 0; i < 8; ++i) { v[i] = expf(v[i] - mx); s += v[i]; }
    red[tid] = s; __syncthreads();
    for (int o = 128; o > 0; o >>= 1) {
        if (tid < o) red[tid] += red[tid + o];
        __syncthreads();
    }
    const float inv = 1.0f / red[0];
#pragma unroll
    for (int i = 0; i < 8; ++i) row[tid + 256 * i] = v[i] * inv;
}

// ---------------------------------------------------------------------------
// f32 -> f16 conversion (for weights)
// ---------------------------------------------------------------------------
__global__ void cvt_kernel(const float* __restrict__ in,
                           _Float16* __restrict__ out, int n) {
    const int i = blockIdx.x * 256 + threadIdx.x;
    if (i < n) out[i] = (_Float16)in[i];
}

// ---------------------------------------------------------------------------
extern "C" void kernel_launch(void* const* d_in, const int* in_sizes, int n_in,
                              void* d_out, int out_size, void* d_ws, size_t ws_size,
                              hipStream_t stream) {
    (void)in_sizes; (void)n_in; (void)out_size; (void)ws_size;

    const float* x      = (const float*)d_in[0];
    const float* gamma1 = (const float*)d_in[1];
    const float* beta1  = (const float*)d_in[2];
    const float* Wq     = (const float*)d_in[3];
    const float* bq     = (const float*)d_in[4];
    const float* Wk     = (const float*)d_in[5];
    const float* bk     = (const float*)d_in[6];
    const float* Wv     = (const float*)d_in[7];
    const float* bv     = (const float*)d_in[8];
    const float* Wo     = (const float*)d_in[9];
    const float* bo     = (const float*)d_in[10];
    const float* gamma2 = (const float*)d_in[11];
    const float* beta2  = (const float*)d_in[12];
    const float* W1     = (const float*)d_in[13];
    const float* b1     = (const float*)d_in[14];
    const float* W2     = (const float*)d_in[15];
    const float* b2     = (const float*)d_in[16];

    char* ws = (char*)d_ws;
    const size_t MB = 1024 * 1024;
    _Float16* wq16 = (_Float16*)(ws + 0 * MB);
    _Float16* wk16 = (_Float16*)(ws + 2 * MB);
    _Float16* wv16 = (_Float16*)(ws + 4 * MB);
    _Float16* wo16 = (_Float16*)(ws + 6 * MB);
    _Float16* w116 = (_Float16*)(ws + 8 * MB);    // 8 MB
    _Float16* w216 = (_Float16*)(ws + 16 * MB);   // 8 MB
    _Float16* nx16 = (_Float16*)(ws + 24 * MB);   // 8 MB (reused for nx2)
    _Float16* q16  = (_Float16*)(ws + 32 * MB);   // 8 MB
    _Float16* kT16 = (_Float16*)(ws + 40 * MB);   // 8 MB
    _Float16* v16b = (_Float16*)(ws + 48 * MB);   // 8 MB
    _Float16* c16  = (_Float16*)(ws + 56 * MB);   // 8 MB
    float*    x1   = (float*)   (ws + 64 * MB);   // 16 MB
    _Float16* h16  = (_Float16*)(ws + 32 * MB);   // 32 MB, reuses q/kT/v/ctx

    float* out_x    = (float*)d_out;
    float* out_attn = out_x + (size_t)TT * DM;

    // --- weight conversions ------------------------------------------------
    const int nDD = DM * DM, nDF = DM * DFF;
    cvt_kernel<<<(nDD + 255) / 256, 256, 0, stream>>>(Wq, wq16, nDD);
    cvt_kernel<<<(nDD + 255) / 256, 256, 0, stream>>>(Wk, wk16, nDD);
    cvt_kernel<<<(nDD + 255) / 256, 256, 0, stream>>>(Wv, wv16, nDD);
    cvt_kernel<<<(nDD + 255) / 256, 256, 0, stream>>>(Wo, wo16, nDD);
    cvt_kernel<<<(nDF + 255) / 256, 256, 0, stream>>>(W1, w116, nDF);
    cvt_kernel<<<(nDF + 255) / 256, 256, 0, stream>>>(W2, w216, nDF);

    // --- LN1 ---------------------------------------------------------------
    layernorm_kernel<<<TT, 256, 0, stream>>>(x, gamma1, beta1, nx16);

    // --- QKV projections ---------------------------------------------------
    dim3 gP(DM / 64, TT / 128, 1);
    gemm_kernel<0, false, false><<<gP, 256, 0, stream>>>(nx16, DM, wq16, DM, q16,  DM, bq, nullptr, DM);
    gemm_kernel<2, false, false><<<gP, 256, 0, stream>>>(nx16, DM, wk16, DM, kT16, DM, bk, nullptr, DM);
    gemm_kernel<0, false, false><<<gP, 256, 0, stream>>>(nx16, DM, wv16, DM, v16b, DM, bv, nullptr, DM);

    // --- attention ---------------------------------------------------------
    dim3 gS(SQ / 64, SQ / 128, BB * NH);
    scores_kernel<<<gS, 256, 0, stream>>>(q16, kT16, out_attn, 0.125f);
    softmax_kernel<<<BB * NH * SQ, 256, 0, stream>>>(out_attn);
    dim3 gC(DK / 64, SQ / 128, BB * NH);
    ctx_kernel<<<gC, 256, 0, stream>>>(out_attn, v16b, c16);

    // --- output projection + residual -> x1 --------------------------------
    gemm_kernel<1, false, true><<<gP, 256, 0, stream>>>(c16, DM, wo16, DM, x1, DM, bo, x, DM);

    // --- LN2 + FFN ---------------------------------------------------------
    layernorm_kernel<<<TT, 256, 0, stream>>>(x1, gamma2, beta2, nx16);
    dim3 gF1(DFF / 64, TT / 128, 1);
    gemm_kernel<0, true,  false><<<gF1, 256, 0, stream>>>(nx16, DM, w116, DFF, h16, DFF, b1, nullptr, DM);
    dim3 gF2(DM / 64, TT / 128, 1);
    gemm_kernel<1, false, true ><<<gF2, 256, 0, stream>>>(h16, DFF, w216, DM, out_x, DM, b2, x1, DFF);
}
